// AdditiveAttention_86019605004512
// MI455X (gfx1250) — matrix operand order repaired and checked by reading, not compiled, not run on hardware
//
#include <hip/hip_runtime.h>
#include <stdint.h>

typedef __attribute__((ext_vector_type(16))) _Float16 v16h;
typedef __attribute__((ext_vector_type(8)))  _Float16 v8h;
typedef __attribute__((ext_vector_type(8)))  float    v8f;

#define BATCH 32
#define NK    4096
#define KD    512
#define HD    256
#define MT    128   // key rows per workgroup
#define KT    32    // K per WMMA step
#define NSTEP (KD / KT)

#if __has_builtin(__builtin_amdgcn_tanhf)
__device__ __forceinline__ float fast_tanh(float x) { return __builtin_amdgcn_tanhf(x); }
#elif __has_builtin(__builtin_amdgcn_tanh_f32)
__device__ __forceinline__ float fast_tanh(float x) { return __builtin_amdgcn_tanh_f32(x); }
#else
__device__ __forceinline__ float fast_tanh(float x) { return tanhf(x); }
#endif

// ---------------------------------------------------------------------------
// Prep: Wt[h][k] = (f16) W_k[k][h]   (256 x 512 f16, stays resident in L2)
// ---------------------------------------------------------------------------
__global__ __launch_bounds__(256) void wt_kernel(const float* __restrict__ Wk,
                                                 _Float16* __restrict__ Wt) {
    int idx = blockIdx.x * 256 + threadIdx.x;   // 0 .. 131071
    int h = idx >> 9;
    int k = idx & 511;
    Wt[idx] = (_Float16)Wk[k * HD + h];
}

// ---------------------------------------------------------------------------
// Prep: qproj[b][h] = sum_d queries[b][d] * W_q[d][h]   (tiny)
// ---------------------------------------------------------------------------
__global__ __launch_bounds__(256) void qproj_kernel(const float* __restrict__ queries,
                                                    const float* __restrict__ Wq,
                                                    float* __restrict__ qproj) {
    int b = blockIdx.x;
    int h = threadIdx.x;
    const float* qrow = queries + (size_t)b * KD;
    float acc = 0.f;
    for (int d = 0; d < KD; ++d)
        acc = fmaf(qrow[d], Wq[d * HD + h], acc);
    qproj[b * HD + h] = acc;
}

// ---------------------------------------------------------------------------
// Main: scores[b][m] = sum_h w_v[h] * tanh( (keys[b]@W_k)[m][h] + qproj[b][h] )
// One workgroup = (batch b, 128 key rows). 8 waves, each wave: 16 rows x 256 cols.
// keys streamed fp32 from HBM via double-buffered async-to-LDS DMA, converted
// to f16 in-register; 16 x v_wmma_f32_16x16x32_f16 per K-step per wave.
// ---------------------------------------------------------------------------
__global__ __launch_bounds__(256) void scores_kernel(const float* __restrict__ keys,
                                                     const _Float16* __restrict__ Wt,
                                                     const float* __restrict__ qproj,
                                                     const float* __restrict__ wv,
                                                     float* __restrict__ scores) {
    // double-buffered; padded rows (multiples of 16B) to spread LDS banks
    __shared__ float    ldsA[2][MT][36];   // 2 x 128 x 32 fp32 (144B rows)
    __shared__ _Float16 ldsB[2][HD][40];   // 2 x 256 x 32 f16  (80B rows)

    const int b    = blockIdx.y;
    const int m0   = blockIdx.x * MT;
    const int tid  = threadIdx.x;
    const int wave = tid >> 5;
    const int lane = tid & 31;
    const int half = lane >> 4;   // 0: lanes 0-15, 1: lanes 16-31
    const int ln   = lane & 15;

    const float* keysB = keys + ((size_t)b * NK + m0) * KD;

    // issue one K-step's staging (8 async b128 per thread = 32KB per WG)
    auto issue_stage = [&](int buf, int k0) {
        #pragma unroll
        for (int i = 0; i < 4; ++i) {            // keys tile: 128x32 fp32 = 16KB
            int c = i * 256 + tid;               // 1024 chunks of 16B
            int row = c >> 3, q = c & 7;
            unsigned lds_off = (unsigned)(uintptr_t)&ldsA[buf][row][q * 4];
            unsigned goff    = (unsigned)((row * KD + k0 + q * 4) * 4);
            asm volatile("global_load_async_to_lds_b128 %0, %1, %2"
                         :: "v"(lds_off), "v"(goff), "s"(keysB) : "memory");
        }
        #pragma unroll
        for (int i = 0; i < 4; ++i) {            // Wt tile: 256x32 f16 = 16KB
            int c = i * 256 + tid;
            int row = c >> 2, p = c & 3;
            unsigned lds_off = (unsigned)(uintptr_t)&ldsB[buf][row][p * 8];
            unsigned goff    = (unsigned)((row * KD + k0) * 2 + p * 16);
            asm volatile("global_load_async_to_lds_b128 %0, %1, %2"
                         :: "v"(lds_off), "v"(goff), "s"(Wt) : "memory");
        }
    };

    v8f acc[16];
    #pragma unroll
    for (int n = 0; n < 16; ++n) acc[n] = (v8f){0.f,0.f,0.f,0.f,0.f,0.f,0.f,0.f};

    issue_stage(0, 0);   // prologue: fill buffer 0

    #pragma unroll 1
    for (int kt = 0; kt < NSTEP; ++kt) {
        const int cur = kt & 1;

        // complete this wave's DMA into buf[cur], then make it WG-visible.
        // the barrier also guarantees everyone finished reading buf[cur^1]
        // last iteration, so we may immediately refill it below.
        asm volatile("s_wait_asynccnt 0" ::: "memory");
        __syncthreads();

        if (kt + 1 < NSTEP)
            issue_stage(cur ^ 1, (kt + 1) * KT);   // overlap DMA with WMMAs

        // ---- A fragment: row = wave*16+ln; half picks K {0..7,16..23} / {8..15,24..31}
        const char*  arow = (const char*)&ldsA[cur][wave * 16 + ln][0];
        const float* af0  = (const float*)(arow + half * 32);        // K base .. base+7
        const float* af1  = (const float*)(arow + half * 32 + 64);   // K base+16 .. base+23
        v16h a;
        #pragma unroll
        for (int j = 0; j < 8; ++j) {
            a[j]     = (_Float16)af0[j];
            a[8 + j] = (_Float16)af1[j];
        }

        // ---- 16 N-tiles of WMMA ----
        #pragma unroll
        for (int n = 0; n < 16; ++n) {
            const char* brow = (const char*)&ldsB[cur][n * 16 + ln][0] + half * 16;
            v8h b0 = *(const v8h*)brow;
            v8h b1 = *(const v8h*)(brow + 32);
            v16h bf;
            #pragma unroll
            for (int j = 0; j < 8; ++j) { bf[j] = b0[j]; bf[8 + j] = b1[j]; }
            acc[n] = __builtin_amdgcn_wmma_f32_16x16x32_f16(
                false, a, false, bf, (short)0, acc[n], false, false);
        }
    }

    // ---- epilogue: score[m] = sum_h wv[h]*tanh(kproj[m][h] + qproj[b][h]) ----
    // C layout: lane holds M = half*8+v (v = vgpr idx), N = n*16+ln
    const float* qb = qproj + b * HD;
    float part[8];
    #pragma unroll
    for (int v = 0; v < 8; ++v) part[v] = 0.f;
    #pragma unroll
    for (int n = 0; n < 16; ++n) {
        int col  = n * 16 + ln;
        float qc = qb[col];
        float wc = wv[col];
        #pragma unroll
        for (int v = 0; v < 8; ++v)
            part[v] = fmaf(wc, fast_tanh(acc[n][v] + qc), part[v]);
    }
    // reduce over the 16 lanes of each half (they cover all 256 columns)
    #pragma unroll
    for (int v = 0; v < 8; ++v) {
        part[v] += __shfl_xor(part[v], 1, 32);
        part[v] += __shfl_xor(part[v], 2, 32);
        part[v] += __shfl_xor(part[v], 4, 32);
        part[v] += __shfl_xor(part[v], 8, 32);
    }
    int rowbase = m0 + wave * 16 + half * 8;
    #pragma unroll
    for (int v = 0; v < 8; ++v)
        if (ln == v) scores[(size_t)b * NK + rowbase + v] = part[v];
}

// ---------------------------------------------------------------------------
// Softmax over NK per batch, times values
// ---------------------------------------------------------------------------
__global__ __launch_bounds__(256) void softmax_kernel(const float* __restrict__ scores,
                                                      const float* __restrict__ values,
                                                      float* __restrict__ out) {
    __shared__ float red[256];
    int b = blockIdx.x, tid = threadIdx.x;
    const float* s = scores + (size_t)b * NK;
    float loc[16];
    float mx = -3.0e38f;
    #pragma unroll
    for (int i = 0; i < 16; ++i) { loc[i] = s[i * 256 + tid]; mx = fmaxf(mx, loc[i]); }
    red[tid] = mx; __syncthreads();
    for (int off = 128; off > 0; off >>= 1) {
        if (tid < off) red[tid] = fmaxf(red[tid], red[tid + off]);
        __syncthreads();
    }
    mx = red[0]; __syncthreads();
    float sum = 0.f;
    #pragma unroll
    for (int i = 0; i < 16; ++i) {
        loc[i] = __expf(loc[i] - mx);
        sum += loc[i];
    }
    red[tid] = sum; __syncthreads();
    for (int off = 128; off > 0; off >>= 1) {
        if (tid < off) red[tid] += red[tid + off];
        __syncthreads();
    }
    float inv = 1.f / red[0];
    #pragma unroll
    for (int i = 0; i < 16; ++i) {
        int k = i * 256 + tid;
        out[(size_t)b * NK + k] = loc[i] * inv * values[(size_t)b * NK + k];
    }
}

// ---------------------------------------------------------------------------
extern "C" void kernel_launch(void* const* d_in, const int* in_sizes, int n_in,
                              void* d_out, int out_size, void* d_ws, size_t ws_size,
                              hipStream_t stream) {
    const float* queries = (const float*)d_in[0];
    const float* keys    = (const float*)d_in[1];
    const float* values  = (const float*)d_in[2];
    const float* Wq      = (const float*)d_in[3];
    const float* Wk      = (const float*)d_in[4];
    const float* wv      = (const float*)d_in[5];
    float* out = (float*)d_out;

    char* ws = (char*)d_ws;
    float*    qproj  = (float*)ws;                       //  32 KB
    _Float16* Wt     = (_Float16*)(ws + 32768);          // 256 KB
    float*    scores = (float*)(ws + 32768 + 262144);    // 512 KB

    wt_kernel   <<<512, 256, 0, stream>>>(Wk, Wt);
    qproj_kernel<<<BATCH, 256, 0, stream>>>(queries, Wq, qproj);
    scores_kernel<<<dim3(NK / MT, BATCH), 256, 0, stream>>>(keys, Wt, qproj, wv, scores);
    softmax_kernel<<<BATCH, 256, 0, stream>>>(scores, values, out);
}
